// GlobalLocalAttention_10307921511021
// MI455X (gfx1250) — matrix operand order repaired and checked by reading, not compiled, not run on hardware
//
#include <hip/hip_runtime.h>

#define BATCH 4
#define CDIM  256
#define HWDIM 4096

typedef unsigned short u16;
typedef unsigned int   u32;

typedef __attribute__((ext_vector_type(16))) __bf16 v16bf;
typedef __attribute__((ext_vector_type(8)))  float  v8f;
typedef __attribute__((ext_vector_type(4)))  u32    u32x4;

union ABReg { u32 u[8]; u32x4 q[2]; v16bf v; };

__device__ __forceinline__ u16 f2bf(float f) {
  u32 u = __float_as_uint(f);
  u32 r = u + 0x7FFFu + ((u >> 16) & 1u);   // round-to-nearest-even
  return (u16)(r >> 16);
}
__device__ __forceinline__ u32 pack2bf(float a, float b) {
  return (u32)f2bf(a) | ((u32)f2bf(b) << 16);
}
__device__ __forceinline__ v8f wmma_bf16(const ABReg& a, const ABReg& b, v8f c) {
  // D = A(16x32 bf16) * B(32x16 bf16) + C(16x16 f32)
  return __builtin_amdgcn_wmma_f32_16x16x32_bf16(
      false, a.v, false, b.v, (short)0, c, false, false);
}

// ---------------------------------------------------------------------------
// Depthwise 3x3 SAME conv. Optional f32 output [B,C,HW]; optional bf16 output
// either [B,C,HW] (bf_transposed=0) or [B,HW,C] (bf_transposed=1).
// ---------------------------------------------------------------------------
__global__ void dwconv3x3_kernel(const float* __restrict__ X,
                                 const float* __restrict__ Wt,
                                 const float* __restrict__ bias,
                                 float* __restrict__ out_f32,
                                 u16* __restrict__ out_bf,
                                 int bf_transposed, int total) {
  int idx = blockIdx.x * blockDim.x + threadIdx.x;
  if (idx >= total) return;
  int w = idx & 63;
  int h = (idx >> 6) & 63;
  int c = (idx >> 12) & 255;
  int b = idx >> 20;
  const float* xc = X + ((size_t)(b * CDIM + c) << 12);
  const float* wk = Wt + c * 9;
  float s = bias[c];
#pragma unroll
  for (int dy = -1; dy <= 1; ++dy) {
    int hh = h + dy;
    if (hh < 0 || hh > 63) continue;
#pragma unroll
    for (int dx = -1; dx <= 1; ++dx) {
      int ww = w + dx;
      if (ww < 0 || ww > 63) continue;
      s += xc[hh * 64 + ww] * wk[(dy + 1) * 3 + (dx + 1)];
    }
  }
  if (out_f32) out_f32[idx] = s;
  if (out_bf) {
    if (bf_transposed)
      out_bf[((size_t)b * HWDIM + (h * 64 + w)) * CDIM + c] = f2bf(s);
    else
      out_bf[idx] = f2bf(s);
  }
}

// ---------------------------------------------------------------------------
// WMMA GEMM: Y[b, hw, o] = sum_c X[b, c, hw] * W[o, c] + bias[o]
// One wave per 16(hw) x 16(o) tile. Weight (B operand) loads are float4.
// ---------------------------------------------------------------------------
__device__ __forceinline__ v8f gemm16_core(const float* __restrict__ Xb,
                                           const float* __restrict__ W,
                                           int C_in, int hwbase, int obase,
                                           int lo, int hi) {
  v8f acc = {0, 0, 0, 0, 0, 0, 0, 0};
  for (int c0 = 0; c0 < C_in; c0 += 32) {
    ABReg a, bm;
#pragma unroll
    for (int j = 0; j < 8; ++j) {
      int k = c0 + 2 * j + 16 * hi;                        // A layout: k = 16*hi + s
      float x0 = Xb[(size_t)k * HWDIM + hwbase + lo];
      float x1 = Xb[(size_t)(k + 1) * HWDIM + hwbase + lo];
      a.u[j] = pack2bf(x0, x1);
    }
    // B operand: 16 contiguous weights, 16B-aligned -> 4x float4
    const float4* wr =
        (const float4*)(W + (size_t)(obase + lo) * C_in + c0 + 16 * hi);
    float4 w0 = wr[0], w1 = wr[1], w2 = wr[2], w3 = wr[3];
    bm.u[0] = pack2bf(w0.x, w0.y);
    bm.u[1] = pack2bf(w0.z, w0.w);
    bm.u[2] = pack2bf(w1.x, w1.y);
    bm.u[3] = pack2bf(w1.z, w1.w);
    bm.u[4] = pack2bf(w2.x, w2.y);
    bm.u[5] = pack2bf(w2.z, w2.w);
    bm.u[6] = pack2bf(w3.x, w3.y);
    bm.u[7] = pack2bf(w3.z, w3.w);
    acc = wmma_bf16(a, bm, acc);
  }
  return acc;
}

__global__ void __launch_bounds__(32)
gemm16_bf16out(const float* __restrict__ X, const float* __restrict__ W,
               const float* __restrict__ bias, u16* __restrict__ Y,
               int C_in, int C_out) {
  const int lane = threadIdx.x;
  const int lo = lane & 15, hi = lane >> 4;
  const int hwbase = blockIdx.x * 16;
  const int obase  = blockIdx.y * 16;
  const int b      = blockIdx.z;
  v8f acc = gemm16_core(X + (size_t)b * C_in * HWDIM, W, C_in, hwbase, obase, lo, hi);
  float bvv = bias[obase + lo];
  u16* yrow = Y + ((size_t)b * HWDIM + hwbase + 8 * hi) * C_out + obase + lo;
#pragma unroll
  for (int r = 0; r < 8; ++r)
    yrow[(size_t)r * C_out] = f2bf(acc[r] + bvv);
}

// f32-out variant stores [B, O, HW]; act: 0=none, 1=relu, 2=sigmoid;
// optional residual (same [B,O,HW] layout) added after activation.
__global__ void __launch_bounds__(32)
gemm16_f32out(const float* __restrict__ X, const float* __restrict__ W,
              const float* __restrict__ bias, float* __restrict__ Y,
              const float* __restrict__ residual, int C_in, int C_out, int act) {
  const int lane = threadIdx.x;
  const int lo = lane & 15, hi = lane >> 4;
  const int hwbase = blockIdx.x * 16;
  const int obase  = blockIdx.y * 16;
  const int b      = blockIdx.z;
  v8f acc = gemm16_core(X + (size_t)b * C_in * HWDIM, W, C_in, hwbase, obase, lo, hi);
  float bvv = bias[obase + lo];
  size_t base = ((size_t)b * C_out + obase + lo) * HWDIM + hwbase + 8 * hi;
#pragma unroll
  for (int r = 0; r < 8; ++r) {
    float v = acc[r] + bvv;
    if (act == 1) v = fmaxf(v, 0.0f);
    else if (act == 2) v = 1.0f / (1.0f + __expf(-v));
    if (residual) v += residual[base + r];
    Y[base + r] = v;
  }
}

// ---------------------------------------------------------------------------
// Flash attention. One wave owns 16 query rows x all 256 channels.
// qbf: [B,HW,C] bf16, kbf: [B,HW,C] bf16 (K^T), vbf: [B,C,HW] bf16.
// gout: [B,C,HW] f32. All operand fetches are 128-bit.
// ---------------------------------------------------------------------------
__global__ void __launch_bounds__(32)
flash_attn_kernel(const u16* __restrict__ qbf, const u16* __restrict__ kbf,
                  const u16* __restrict__ vbf, float* __restrict__ gout) {
  __shared__ alignas(16) u16 lds_p[16 * 32];
  const int lane = threadIdx.x;
  const int lo = lane & 15, hi = lane >> 4;
  const int b = blockIdx.y;
  const int qbase = blockIdx.x * 16;

  const u16* qb = qbf + (size_t)b * HWDIM * CDIM;
  const u16* kb = kbf + (size_t)b * HWDIM * CDIM;
  const u16* vb = vbf + (size_t)b * CDIM * HWDIM;

  // Q tile, resident in A layout. Per lane: two 16B chunks per 32-channel step.
  ABReg qa[8];
  {
    const u16* qrow = qb + (size_t)(qbase + lo) * CDIM;
#pragma unroll
    for (int kk = 0; kk < 8; ++kk) {
      const u16* p = qrow + kk * 32 + 16 * hi;
      qa[kk].q[0] = *(const u32x4*)(p);
      qa[kk].q[1] = *(const u32x4*)(p + 8);
    }
  }

  v8f oacc[16];
#pragma unroll
  for (int t = 0; t < 16; ++t) oacc[t] = (v8f){0, 0, 0, 0, 0, 0, 0, 0};
  float m_run[8], l_run[8];
#pragma unroll
  for (int r = 0; r < 8; ++r) { m_run[r] = -1e30f; l_run[r] = 0.0f; }
  const float inv_scale = 1.0f / 16.0f;   // 1/sqrt(C)

  for (int kb0 = 0; kb0 < HWDIM; kb0 += 32) {
    // Prefetch next key block's K^T rows (one row per lane) into cache.
    if (kb0 + 32 < HWDIM)
      __builtin_prefetch(kb + (size_t)(kb0 + 32 + lane) * CDIM, 0, 3);

    // S = Q * K^T for 32 keys (two 16-key N sub-tiles)
    v8f s[2];
#pragma unroll
    for (int t = 0; t < 2; ++t) {
      v8f acc = {0, 0, 0, 0, 0, 0, 0, 0};
      const u16* krow = kb + (size_t)(kb0 + 16 * t + lo) * CDIM + 16 * hi;
#pragma unroll
      for (int kk = 0; kk < 8; ++kk) {
        ABReg bm;
        const u32x4* kp = (const u32x4*)(krow + kk * 32);
        bm.q[0] = kp[0];
        bm.q[1] = kp[1];
        acc = wmma_bf16(qa[kk], bm, acc);
      }
      s[t] = acc;
    }

    // Online softmax. Row m = r + 8*hi lives across the 16 lanes of a half-wave.
    float pvv[2][8], alpha[8];
#pragma unroll
    for (int r = 0; r < 8; ++r) {
      float mx = fmaxf(s[0][r], s[1][r]) * inv_scale;
      for (int d = 1; d < 16; d <<= 1) mx = fmaxf(mx, __shfl_xor(mx, d, 16));
      float mnew = fmaxf(m_run[r], mx);
      float a = __expf(m_run[r] - mnew);
      m_run[r] = mnew;
      alpha[r] = a;
      float p0 = __expf(s[0][r] * inv_scale - mnew);
      float p1 = __expf(s[1][r] * inv_scale - mnew);
      pvv[0][r] = p0; pvv[1][r] = p1;
      float rs = p0 + p1;
      for (int d = 1; d < 16; d <<= 1) rs += __shfl_xor(rs, d, 16);
      l_run[r] = l_run[r] * a + rs;
    }
#pragma unroll
    for (int t = 0; t < 16; ++t)
#pragma unroll
      for (int r = 0; r < 8; ++r) oacc[t][r] *= alpha[r];

    // P: C/D layout -> LDS -> A layout (single-wave WG, barrier is free)
    __syncthreads();
#pragma unroll
    for (int t = 0; t < 2; ++t)
#pragma unroll
      for (int r = 0; r < 8; ++r)
        lds_p[(r + 8 * hi) * 32 + 16 * t + lo] = f2bf(pvv[t][r]);
    __syncthreads();
    ABReg pa;
    {
      const u16* prow = lds_p + lo * 32 + 16 * hi;
      pa.q[0] = *(const u32x4*)(prow);         // ds_load_b128
      pa.q[1] = *(const u32x4*)(prow + 8);
    }

    // O += P * V : 16 channel tiles, 32B of V per lane per tile
#pragma unroll
    for (int ct = 0; ct < 16; ++ct) {
      ABReg bvm;
      const u32x4* vp =
          (const u32x4*)(vb + (size_t)(ct * 16 + lo) * HWDIM + kb0 + 16 * hi);
      bvm.q[0] = vp[0];
      bvm.q[1] = vp[1];
      oacc[ct] = wmma_bf16(pa, bvm, oacc[ct]);
    }
  }

  // Normalize and store as [B, C, HW]
#pragma unroll
  for (int r = 0; r < 8; ++r) {
    float inv_l = 1.0f / l_run[r];
    int m = qbase + r + 8 * hi;
#pragma unroll
    for (int ct = 0; ct < 16; ++ct)
      gout[(size_t)b * CDIM * HWDIM + (size_t)(ct * 16 + lo) * HWDIM + m] =
          oacc[ct][r] * inv_l;
  }
}

// ---------------------------------------------------------------------------
// GAP: one block per (b,c) row, coalesced, LDS tree reduce.
// ---------------------------------------------------------------------------
__global__ void __launch_bounds__(256)
gap_reduce_kernel(const float* __restrict__ gout, float* __restrict__ gap) {
  __shared__ float red[256];
  int row = blockIdx.x;                       // b*C + c
  int t = threadIdx.x;
  const float* p = gout + (size_t)row * HWDIM;
  float s = 0.0f;
#pragma unroll
  for (int i = 0; i < HWDIM / 256; ++i) s += p[t + i * 256];
  red[t] = s;
  __syncthreads();
  for (int d = 128; d > 0; d >>= 1) {
    if (t < d) red[t] += red[t + d];
    __syncthreads();
  }
  if (t == 0) gap[row] = red[0] * (1.0f / HWDIM);
}

// Channel squeeze-excite: cw[b,c] = sigmoid(Wc2 relu(Wc1 gap + bc1) + bc2)
__global__ void __launch_bounds__(256)
cw_kernel(const float* __restrict__ gap, const float* __restrict__ Wc1,
          const float* __restrict__ bc1, const float* __restrict__ Wc2,
          const float* __restrict__ bc2, float* __restrict__ cw) {
  __shared__ float g[CDIM];
  __shared__ float h1[32];
  int b = blockIdx.x;
  int c = threadIdx.x;
  g[c] = gap[b * CDIM + c];
  __syncthreads();
  if (c < 32) {
    float a = bc1[c];
    for (int j = 0; j < CDIM; ++j) a += Wc1[c * CDIM + j] * g[j];
    h1[c] = fmaxf(a, 0.0f);
  }
  __syncthreads();
  float a = bc2[c];
  for (int j = 0; j < 32; ++j) a += Wc2[c * 32 + j] * h1[j];
  cw[b * CDIM + c] = 1.0f / (1.0f + __expf(-a));
}

// fused = global*cw + local*spatial   (all [B,C,HW] f32; cw broadcast per channel)
__global__ void fuse_kernel(const float* __restrict__ g, const float* __restrict__ l,
                            const float* __restrict__ sw, const float* __restrict__ cw,
                            float* __restrict__ fused, int total) {
  int i = blockIdx.x * blockDim.x + threadIdx.x;
  if (i >= total) return;
  int bc = i >> 12;           // i / HW == b*C + c
  fused[i] = g[i] * cw[bc] + l[i] * sw[i];
}

// ---------------------------------------------------------------------------
extern "C" void kernel_launch(void* const* d_in, const int* in_sizes, int n_in,
                              void* d_out, int out_size, void* d_ws, size_t ws_size,
                              hipStream_t stream) {
  const float* x   = (const float*)d_in[0];
  const float* Wq  = (const float*)d_in[1];
  const float* bq  = (const float*)d_in[2];
  const float* Wk  = (const float*)d_in[3];
  const float* bk  = (const float*)d_in[4];
  const float* Wv  = (const float*)d_in[5];
  const float* bv  = (const float*)d_in[6];
  const float* Wl  = (const float*)d_in[7];
  const float* bl  = (const float*)d_in[8];
  const float* Ws1 = (const float*)d_in[9];
  const float* bs1 = (const float*)d_in[10];
  const float* Ws2 = (const float*)d_in[11];
  const float* bs2 = (const float*)d_in[12];
  const float* Wc1 = (const float*)d_in[13];
  const float* bc1 = (const float*)d_in[14];
  const float* Wc2 = (const float*)d_in[15];
  const float* bc2 = (const float*)d_in[16];
  const float* Wo  = (const float*)d_in[17];
  const float* bo  = (const float*)d_in[18];
  float* out = (float*)d_out;

  char* ws = (char*)d_ws;
  size_t off = 0;
  auto alloc = [&](size_t bytes) -> void* {
    void* p = ws + off;
    off = (off + bytes + 255) & ~(size_t)255;
    return p;
  };
  const size_t nBCHW = (size_t)BATCH * CDIM * HWDIM;          // 4M elems
  u16*   q_bf  = (u16*)  alloc(nBCHW * 2);                    // [B,HW,C]
  u16*   k_bf  = (u16*)  alloc(nBCHW * 2);                    // [B,HW,C]
  u16*   v_bf  = (u16*)  alloc(nBCHW * 2);                    // [B,C,HW]
  float* v_img = (float*)alloc(nBCHW * 4);                    // reused as `fused`
  float* gouts = (float*)alloc(nBCHW * 4);
  float* louts = (float*)alloc(nBCHW * 4);
  float* s1    = (float*)alloc((size_t)BATCH * 64 * HWDIM * 4);
  float* swb   = (float*)alloc(nBCHW * 4);
  float* gapb  = (float*)alloc((size_t)BATCH * CDIM * 4);
  float* cw    = (float*)alloc((size_t)BATCH * CDIM * 4);
  float* fused = v_img;                                       // v_img dead after local conv

  const int total = (int)nBCHW;
  const dim3 ew((total + 255) / 256), eb(256);
  const dim3 g16(HWDIM / 16, CDIM / 16, BATCH), w32(32);

  // k = dw3x3(x; Wk) -> bf16 [B,HW,C]   (K^T layout for WMMA B operand)
  dwconv3x3_kernel<<<ew, eb, 0, stream>>>(x, Wk, bk, nullptr, k_bf, 1, total);
  // v = dw3x3(x; Wv) -> f32 [B,C,HW] + bf16 [B,C,HW]
  dwconv3x3_kernel<<<ew, eb, 0, stream>>>(x, Wv, bv, v_img, v_bf, 0, total);
  // q = Wq x + bq -> bf16 [B,HW,C]
  gemm16_bf16out<<<g16, w32, 0, stream>>>(x, Wq, bq, q_bf, CDIM, CDIM);
  // global_out = softmax(QK^T/16) V -> f32 [B,C,HW]
  flash_attn_kernel<<<dim3(HWDIM / 16, BATCH), w32, 0, stream>>>(q_bf, k_bf, v_bf, gouts);
  // local_out = dw3x3(v_img; Wl)
  dwconv3x3_kernel<<<ew, eb, 0, stream>>>(v_img, Wl, bl, louts, nullptr, 0, total);
  // spatial attention: sigmoid(Ws2 relu(Ws1 local + bs1) + bs2)
  gemm16_f32out<<<dim3(HWDIM / 16, 64 / 16, BATCH), w32, 0, stream>>>(
      louts, Ws1, bs1, s1, nullptr, CDIM, 64, 1);
  gemm16_f32out<<<g16, w32, 0, stream>>>(s1, Ws2, bs2, swb, nullptr, 64, CDIM, 2);
  // channel attention from GAP(global_out)
  gap_reduce_kernel<<<BATCH * CDIM, 256, 0, stream>>>(gouts, gapb);
  cw_kernel<<<BATCH, 256, 0, stream>>>(gapb, Wc1, bc1, Wc2, bc2, cw);
  // fused = global*cw + local*spatial   (overwrites v_img buffer)
  fuse_kernel<<<ew, eb, 0, stream>>>(gouts, louts, swb, cw, fused, total);
  // out = Wo fused + bo + x
  gemm16_f32out<<<g16, w32, 0, stream>>>(fused, Wo, bo, out, x, CDIM, CDIM, 0);
}
